// SpatialGNN_19851338842751
// MI455X (gfx1250) — matrix operand order repaired and checked by reading, not compiled, not run on hardware
//
#include <hip/hip_runtime.h>
#include <math.h>

typedef __attribute__((ext_vector_type(16))) __bf16 v16bf;
typedef __attribute__((ext_vector_type(8)))  float  v8f;

#ifndef GNN_N
#define GNN_N 100000
#define GNN_E 640000
#define GNN_B 64
#endif

// ---------------------------------------------------------------- helpers
__device__ __forceinline__ unsigned short f2bf(float f) {
  union { float f; unsigned u; } x; x.f = f;
  unsigned r = x.u + 0x7FFFu + ((x.u >> 16) & 1u);   // round-to-nearest-even
  return (unsigned short)(r >> 16);
}

__device__ __forceinline__ float gelu_erf(float x) {
  return 0.5f * x * (1.0f + erff(x * 0.70710678118654752f));
}

__device__ __forceinline__ void atomicMaxF(float* addr, float val) {
  float old = *((volatile float*)addr);
  while (val > old) {
    int assumed = __float_as_int(old);
    int prev = atomicCAS((int*)addr, assumed, __float_as_int(val));
    if (prev == assumed) break;
    old = __int_as_float(prev);
  }
}

// ---------------------------------------------------------------- fills / converts
__global__ __launch_bounds__(256) void k_fill(float* p, float v, long n) {
  long i = (long)blockIdx.x * blockDim.x + threadIdx.x;
  if (i < n) p[i] = v;
}

__global__ __launch_bounds__(256) void k_cvt_bf16(const float* __restrict__ in,
                                                  unsigned short* __restrict__ out, long n) {
  long i = (long)blockIdx.x * blockDim.x + threadIdx.x;
  if (i < n) out[i] = f2bf(in[i]);
}

// ---------------------------------------------------------------- WMMA GEMM
// Y[M,128] = bf16(X)[M,128] @ bf16(W)[128,128] + bias ; fp32 accumulate.
// 8 waves per block; each wave owns TWO 16-row M-tiles (32 rows) x all 128 cols,
// so every B fragment pulled from LDS feeds two v_wmma ops (halves DS traffic
// per matrix op and hides ds_load latency behind in-flight WMMAs).
__global__ __launch_bounds__(256) void k_wmma_gemm128(
    const unsigned short* __restrict__ Xb,  // [M,128] bf16
    const float* __restrict__ W,            // [128,128] f32 (k-major: W[k*128+n])
    const float* __restrict__ bias,         // [128]
    float* __restrict__ Y,                  // [M,128] f32
    int M) {
  __shared__ unsigned short sW[128 * 128]; // transposed: sW[n*128 + k], bf16

  const int t = threadIdx.x;
  for (int i = t; i < 128 * 128; i += 256) {
    int k = i >> 7, n = i & 127;
    sW[n * 128 + k] = f2bf(W[i]);
  }
  __syncthreads();

  const int wave = t >> 5, lane = t & 31;
  const long row0 = ((long)blockIdx.x * 8 + wave) * 32;
  if (row0 >= M) return;                      // uniform per-wave; EXEC stays all-ones

  const int m   = lane & 15;                  // M row within tile (A) / N col (B,D)
  const int grp = lane >> 4;                  // lane half

  v8f acc0[8], acc1[8];
  const v8f vzero = {0.f, 0.f, 0.f, 0.f, 0.f, 0.f, 0.f, 0.f};
#pragma unroll
  for (int j = 0; j < 8; ++j) { acc0[j] = vzero; acc1[j] = vzero; }

  const unsigned short* Xrow0 = Xb + (row0 + m) * 128;
  const unsigned short* Xrow1 = Xb + (row0 + 16 + m) * 128;

#pragma unroll
  for (int kt = 0; kt < 4; ++kt) {
    // A fragments: VGPR v, half s ->
    //   k = grp*16 + 2v+s (contiguous, matches B)
    union { unsigned u[8]; v16bf v; } af0, af1;
#pragma unroll
    for (int vv = 0; vv < 8; ++vv) {
      int k0 = grp * 16 + vv * 2;
      af0.u[vv] = *reinterpret_cast<const unsigned*>(Xrow0 + kt * 32 + k0);
      af1.u[vv] = *reinterpret_cast<const unsigned*>(Xrow1 + kt * 32 + k0);
    }
    // B fragment 32x16: lane holds col n = j*16+m, 16 consecutive k from grp*16.
    // Software-pipelined: issue ds_loads for fragment j+1 before the two WMMAs
    // on fragment j so the dscnt wait retires behind matrix-pipe work.
    const int kbase = kt * 32 + grp * 16;
    union { uint4 q[2]; v16bf v; } bcur, bnxt;
    {
      const uint4* p = reinterpret_cast<const uint4*>(&sW[(0 * 16 + m) * 128 + kbase]);
      bcur.q[0] = p[0]; bcur.q[1] = p[1];
    }
#pragma unroll
    for (int j = 0; j < 8; ++j) {
      if (j < 7) {
        const uint4* p = reinterpret_cast<const uint4*>(&sW[((j + 1) * 16 + m) * 128 + kbase]);
        bnxt.q[0] = p[0]; bnxt.q[1] = p[1];
      }
      acc0[j] = __builtin_amdgcn_wmma_f32_16x16x32_bf16(
          false, af0.v, false, bcur.v, (short)0, acc0[j], false, false);
      acc1[j] = __builtin_amdgcn_wmma_f32_16x16x32_bf16(
          false, af1.v, false, bcur.v, (short)0, acc1[j], false, false);
      bcur = bnxt;
    }
  }

  // D layout: lane -> col n = j*16+m ; VGPR r -> row grp*8 + r
#pragma unroll
  for (int j = 0; j < 8; ++j) {
    const int n = j * 16 + m;
    const float bn = bias[n];
#pragma unroll
    for (int r = 0; r < 8; ++r) {
      const long row = row0 + grp * 8 + r;
      Y[row * 128 + n]        = acc0[j][r] + bn;
      Y[(row + 16) * 128 + n] = acc1[j][r] + bn;
    }
  }
}

// ---------------------------------------------------------------- small MLPs
// node_mlp stage 1: h = LN(gelu(x@W1+b1)) -> bf16
__global__ __launch_bounds__(128) void k_node_l1(
    const float* __restrict__ x, const float* __restrict__ W, const float* __restrict__ b,
    const float* __restrict__ lng, const float* __restrict__ lnb,
    unsigned short* __restrict__ out) {
  __shared__ float sx[10];
  __shared__ float s1[128], s2[128];
  const int nid = blockIdx.x, t = threadIdx.x;
  if (t < 10) sx[t] = x[(long)nid * 10 + t];
  __syncthreads();
  float acc = b[t];
#pragma unroll
  for (int i = 0; i < 10; ++i) acc += sx[i] * W[i * 128 + t];
  float g = gelu_erf(acc);
  s1[t] = g; s2[t] = g * g;
  __syncthreads();
  for (int off = 64; off > 0; off >>= 1) {
    if (t < off) { s1[t] += s1[t + off]; s2[t] += s2[t + off]; }
    __syncthreads();
  }
  const float mean = s1[0] * (1.f / 128.f);
  const float var  = s2[0] * (1.f / 128.f) - mean * mean;
  const float y = (g - mean) * rsqrtf(var + 1e-5f) * lng[t] + lnb[t];
  out[(long)nid * 128 + t] = f2bf(y);
}

// edge_mlp stage 1: h = gelu(ea@W1+b1) -> bf16
__global__ __launch_bounds__(128) void k_edge_l1(
    const float* __restrict__ ea, const float* __restrict__ W, const float* __restrict__ b,
    unsigned short* __restrict__ out) {
  __shared__ float se[6];
  const int eid = blockIdx.x, t = threadIdx.x;
  if (t < 6) se[t] = ea[(long)eid * 6 + t];
  __syncthreads();
  float acc = b[t];
#pragma unroll
  for (int i = 0; i < 6; ++i) acc += se[i] * W[i * 128 + t];
  out[(long)eid * 128 + t] = f2bf(gelu_erf(acc));
}

// global_mlp: u128 = gelu(u@W1+b1)@W2+b2 (B=64 rows only)
__global__ __launch_bounds__(128) void k_global_mlp(
    const float* __restrict__ u, const float* __restrict__ W1, const float* __restrict__ b1,
    const float* __restrict__ W2, const float* __restrict__ b2, float* __restrict__ out) {
  __shared__ float su[5];
  __shared__ float h[128];
  const int bid = blockIdx.x, t = threadIdx.x;
  if (t < 5) su[t] = u[(long)bid * 5 + t];
  __syncthreads();
  float acc = b1[t];
#pragma unroll
  for (int i = 0; i < 5; ++i) acc += su[i] * W1[i * 128 + t];
  h[t] = gelu_erf(acc);
  __syncthreads();
  float o = b2[t];
  for (int k = 0; k < 128; ++k) o += h[k] * W2[k * 128 + t];
  out[(long)bid * 128 + t] = o;
}

// ---------------------------------------------------------------- attention edge phase
__global__ __launch_bounds__(256) void k_alpha_dot(
    const float* __restrict__ q, const float* __restrict__ kk, const float* __restrict__ eh,
    const int* __restrict__ src, const int* __restrict__ dst,
    float* __restrict__ alphaR, float* __restrict__ amax, long n) {
  long t = (long)blockIdx.x * blockDim.x + threadIdx.x;
  if (t >= n) return;
  const long e = t >> 2; const int h = (int)(t & 3);
  const int s = src[e], d = dst[e];
  const float4* qp = (const float4*)(q  + (long)d * 128 + h * 32);
  const float4* kp = (const float4*)(kk + (long)s * 128 + h * 32);
  const float4* ep = (const float4*)(eh + (long)e * 128 + h * 32);
  float acc = 0.f;
#pragma unroll
  for (int i = 0; i < 8; ++i) {
    float4 a = qp[i], b = kp[i], c = ep[i];
    acc += a.x * (b.x + c.x) + a.y * (b.y + c.y) + a.z * (b.z + c.z) + a.w * (b.w + c.w);
  }
  acc *= 0.1767766952966369f; // 1/sqrt(32)
  alphaR[t] = acc;
  atomicMaxF(&amax[(long)d * 4 + h], acc);
}

__global__ __launch_bounds__(256) void k_alpha_exp(
    const float* __restrict__ alphaR, const float* __restrict__ amax,
    const int* __restrict__ dst, float* __restrict__ alphaE,
    float* __restrict__ asum, long n) {
  long t = (long)blockIdx.x * blockDim.x + threadIdx.x;
  if (t >= n) return;
  const long e = t >> 2; const int h = (int)(t & 3);
  const int d = dst[e];
  const float a = __expf(alphaR[t] - amax[(long)d * 4 + h]);
  alphaE[t] = a;
  atomicAdd(&asum[(long)d * 4 + h], a);
}

__global__ __launch_bounds__(256) void k_message(
    const float* __restrict__ vf, const float* __restrict__ eh,
    const float* __restrict__ alphaE, const float* __restrict__ asum,
    const int* __restrict__ src, const int* __restrict__ dst,
    float* __restrict__ agg, long n) {
  long t = (long)blockIdx.x * blockDim.x + threadIdx.x;
  if (t >= n) return;
  const long e = t >> 5; const int c = (int)(t & 31); const int head = c >> 3;
  const int s = src[e], d = dst[e];
  const float a = alphaE[e * 4 + head] / (asum[(long)d * 4 + head] + 1e-16f);
  const float4 vv = ((const float4*)(vf + (long)s * 128))[c];
  const float4 ee = ((const float4*)(eh + (long)e * 128))[c];
  float* o = agg + (long)d * 128 + c * 4;
  atomicAdd(o + 0, (vv.x + ee.x) * a);
  atomicAdd(o + 1, (vv.y + ee.y) * a);
  atomicAdd(o + 2, (vv.z + ee.z) * a);
  atomicAdd(o + 3, (vv.w + ee.w) * a);
}

// x = LN(x + gelu(agg + skip)) ; also emit bf16 mirror for next layer's GEMMs
__global__ __launch_bounds__(128) void k_node_update(
    float* __restrict__ xf, const float* __restrict__ agg, const float* __restrict__ skf,
    const float* __restrict__ g, const float* __restrict__ bta,
    unsigned short* __restrict__ xb) {
  __shared__ float s1[128], s2[128];
  const int nid = blockIdx.x, t = threadIdx.x;
  const long idx = (long)nid * 128 + t;
  const float xi = agg[idx] + skf[idx];
  const float h = xf[idx] + gelu_erf(xi);
  s1[t] = h; s2[t] = h * h;
  __syncthreads();
  for (int off = 64; off > 0; off >>= 1) {
    if (t < off) { s1[t] += s1[t + off]; s2[t] += s2[t + off]; }
    __syncthreads();
  }
  const float mean = s1[0] * (1.f / 128.f);
  const float var  = s2[0] * (1.f / 128.f) - mean * mean;
  const float y = (h - mean) * rsqrtf(var + 1e-5f) * g[t] + bta[t];
  xf[idx] = y;
  xb[idx] = f2bf(y);
}

__global__ __launch_bounds__(256) void k_final_gate(
    const float* __restrict__ xf, const float* __restrict__ u128,
    const int* __restrict__ batch, float* __restrict__ out, long n) {
  long t = (long)blockIdx.x * blockDim.x + threadIdx.x;
  if (t >= n) return;
  const long node = t >> 7; const int f = (int)(t & 127);
  const float ue = u128[(long)batch[node] * 128 + f];
  const float gg = 1.f / (1.f + __expf(-ue));
  out[t] = xf[t] * gg + ue * (1.f - gg);
}

__global__ __launch_bounds__(256) void k_copy(const float* __restrict__ in,
                                              float* __restrict__ out, long n) {
  long t = (long)blockIdx.x * blockDim.x + threadIdx.x;
  if (t < n) out[t] = in[t];
}

// ---------------------------------------------------------------- host orchestration
extern "C" void kernel_launch(void* const* d_in, const int* in_sizes, int n_in,
                              void* d_out, int out_size, void* d_ws, size_t ws_size,
                              hipStream_t stream) {
  (void)in_sizes; (void)n_in; (void)out_size; (void)ws_size;
  const long N = GNN_N, E = GNN_E, B = GNN_B;

  // Flattened input order (pytree, sorted dict keys):
  // 0 batch | 1 edge_attr | 2 edge_index |
  // 3.. convs[i]: e.W,e.b,k.W,k.b,q.W,q.b,skip.W,skip.b,v.W,v.b (10 each, i=0..2)
  // 33 edge_mlp.l1.W,34 .b,35 .l2.W,36 .b | 37..40 global_mlp | 41..46 node_mlp
  // 47..52 norms[i].b,.g | 53 u | 54 x
  const int*   batch = (const int*)d_in[0];
  const float* eattr = (const float*)d_in[1];
  const int*   src   = (const int*)d_in[2];
  const int*   dst   = src + E;
  struct CW { const float *eW,*ebv,*kW,*kb,*qW,*qb,*sW,*sb,*vW,*vb; } cw[3];
  for (int i = 0; i < 3; ++i) {
    int p = 3 + i * 10;
    cw[i] = { (const float*)d_in[p+0], (const float*)d_in[p+1],
              (const float*)d_in[p+2], (const float*)d_in[p+3],
              (const float*)d_in[p+4], (const float*)d_in[p+5],
              (const float*)d_in[p+6], (const float*)d_in[p+7],
              (const float*)d_in[p+8], (const float*)d_in[p+9] };
  }
  const float* emW1 = (const float*)d_in[33]; const float* emb1 = (const float*)d_in[34];
  const float* emW2 = (const float*)d_in[35]; const float* emb2 = (const float*)d_in[36];
  const float* gmW1 = (const float*)d_in[37]; const float* gmb1 = (const float*)d_in[38];
  const float* gmW2 = (const float*)d_in[39]; const float* gmb2 = (const float*)d_in[40];
  const float* nmW1 = (const float*)d_in[41]; const float* nmb1 = (const float*)d_in[42];
  const float* nmW2 = (const float*)d_in[43]; const float* nmb2 = (const float*)d_in[44];
  const float* lnB  = (const float*)d_in[45]; const float* lnG  = (const float*)d_in[46];
  const float* nrmB[3] = { (const float*)d_in[47], (const float*)d_in[49], (const float*)d_in[51] };
  const float* nrmG[3] = { (const float*)d_in[48], (const float*)d_in[50], (const float*)d_in[52] };
  const float* uin = (const float*)d_in[53];
  const float* xin = (const float*)d_in[54];

  // workspace carve
  char* w = (char*)d_ws;
  auto alloc = [&](size_t bytes) { char* r = w; w += (bytes + 255) & ~(size_t)255; return r; };
  float* xf   = (float*)alloc((size_t)N * 128 * 4);
  float* qf   = (float*)alloc((size_t)N * 128 * 4);
  float* kf   = (float*)alloc((size_t)N * 128 * 4);
  float* vf   = (float*)alloc((size_t)N * 128 * 4);
  float* skf  = (float*)alloc((size_t)N * 128 * 4);
  float* agg  = (float*)alloc((size_t)N * 128 * 4);
  float* ef   = (float*)alloc((size_t)E * 128 * 4);
  float* eh   = (float*)alloc((size_t)E * 128 * 4);
  float* u128 = (float*)alloc((size_t)B * 128 * 4);
  float* aR   = (float*)alloc((size_t)E * 4 * 4);
  float* aE   = (float*)alloc((size_t)E * 4 * 4);
  float* amax = (float*)alloc((size_t)N * 4 * 4);
  float* asum = (float*)alloc((size_t)N * 4 * 4);
  unsigned short* xb = (unsigned short*)alloc((size_t)N * 128 * 2);
  unsigned short* eb = (unsigned short*)alloc((size_t)E * 128 * 2);

  auto cdiv = [](long a, long b) { return (int)((a + b - 1) / b); };
  const int gemmN = cdiv(N, 256), gemmE = cdiv(E, 256);   // 256 rows per block (8 waves x 32)

  // node_mlp: LN(gelu(x@W1+b1)) -> bf16 ; @W2+b2 via WMMA -> xf ; bf16 mirror
  k_node_l1<<<(int)N, 128, 0, stream>>>(xin, nmW1, nmb1, lnG, lnB, xb);
  k_wmma_gemm128<<<gemmN, 256, 0, stream>>>(xb, nmW2, nmb2, xf, (int)N);
  k_cvt_bf16<<<cdiv(N * 128, 256), 256, 0, stream>>>(xf, xb, N * 128);

  // edge_mlp
  k_edge_l1<<<(int)E, 128, 0, stream>>>(eattr, emW1, emb1, eb);
  k_wmma_gemm128<<<gemmE, 256, 0, stream>>>(eb, emW2, emb2, ef, (int)E);
  k_cvt_bf16<<<cdiv(E * 128, 256), 256, 0, stream>>>(ef, eb, E * 128);

  // global_mlp
  k_global_mlp<<<(int)B, 128, 0, stream>>>(uin, gmW1, gmb1, gmW2, gmb2, u128);

  for (int L = 0; L < 3; ++L) {
    k_wmma_gemm128<<<gemmN, 256, 0, stream>>>(xb, cw[L].qW, cw[L].qb, qf,  (int)N);
    k_wmma_gemm128<<<gemmN, 256, 0, stream>>>(xb, cw[L].kW, cw[L].kb, kf,  (int)N);
    k_wmma_gemm128<<<gemmN, 256, 0, stream>>>(xb, cw[L].vW, cw[L].vb, vf,  (int)N);
    k_wmma_gemm128<<<gemmN, 256, 0, stream>>>(xb, cw[L].sW, cw[L].sb, skf, (int)N);
    k_wmma_gemm128<<<gemmE, 256, 0, stream>>>(eb, cw[L].eW, cw[L].ebv, eh, (int)E);

    k_fill<<<cdiv(N * 128, 256), 256, 0, stream>>>(agg, 0.f, N * 128);
    k_fill<<<cdiv(N * 4, 256), 256, 0, stream>>>(amax, -3.0e38f, N * 4);
    k_fill<<<cdiv(N * 4, 256), 256, 0, stream>>>(asum, 0.f, N * 4);

    k_alpha_dot<<<cdiv(E * 4, 256), 256, 0, stream>>>(qf, kf, eh, src, dst, aR, amax, E * 4);
    k_alpha_exp<<<cdiv(E * 4, 256), 256, 0, stream>>>(aR, amax, dst, aE, asum, E * 4);
    k_message<<<cdiv(E * 32, 256), 256, 0, stream>>>(vf, eh, aE, asum, src, dst, agg, E * 32);

    k_node_update<<<(int)N, 128, 0, stream>>>(xf, agg, skf, nrmG[L], nrmB[L], xb);
  }

  float* out = (float*)d_out;
  k_final_gate<<<cdiv(N * 128, 256), 256, 0, stream>>>(xf, u128, batch, out, N * 128);
  k_copy<<<cdiv(B * 128, 256), 256, 0, stream>>>(u128, out + N * 128, B * 128);
}
